// L2MLinear_3805341024601
// MI455X (gfx1250) — matrix operand order repaired and checked by reading, not compiled, not run on hardware
//
#include <hip/hip_runtime.h>
#include <stdint.h>

#define BATCH   32
#define NSEQ    512
#define INDIM   1024
#define OUTDIM  1024

#define BM      128          // block tile rows (N dim)
#define BN      64           // block tile cols (OUT dim)
#define KB      32           // K step = bf16 WMMA depth
#define KSTEPS  (INDIM / KB) // 32

#define NCHUNKS (NSEQ / BM)   // 4
#define JCHUNKS (OUTDIM / BN) // 16

// fragment-major blob sizes (elements) per (tile, K-step)
#define XBLOB   (BM * KB)     // 4096  = 8 subtiles x 32 lanes x 16
#define WBLOB   (KB * BN)     // 2048  = 4 subtiles x 32 lanes x 16

// ---- workspace layout (bytes) for the two-phase fast path
#define WS_XH   0ull
#define WS_XL   33554432ull    // + 32 MiB
#define WS_WH   67108864ull    // + 32 MiB
#define WS_WL   134217728ull   // + 64 MiB
#define WS_NEED 201326592ull   // 192 MiB total

// ---- fused-fallback LDS carve (round-2 kernel)
#define XPITCH  36
#define WPITCH  66
#define SX_BYTES   (2 * BM * XPITCH * 4)
#define SW_BYTES   (2 * KB * WPITCH * 4)
#define XH_BYTES   (BM * KB * 2)
#define WH_BYTES   (KB * BN * 2)
#define OFF_SX     0
#define OFF_SW     (OFF_SX + SX_BYTES)
#define OFF_XH     (OFF_SW + SW_BYTES)
#define OFF_XL     (OFF_XH + XH_BYTES)
#define OFF_WH     (OFF_XL + XH_BYTES)
#define OFF_WL     (OFF_WH + WH_BYTES)
#define SMEM_FUSED (OFF_WL + WH_BYTES)     // 78336 B

// ---- fast-path GEMM LDS carve: double-buffered bf16 fragment blobs
#define FOFF_XH    0
#define FOFF_XL    (FOFF_XH + 2 * XBLOB * 2)   // 16384
#define FOFF_WH    (FOFF_XL + 2 * XBLOB * 2)   // 32768
#define FOFF_WL    (FOFF_WH + 2 * WBLOB * 2)   // 40960
#define SMEM_FAST  (FOFF_WL + 2 * WBLOB * 2)   // 49152 B

typedef __attribute__((ext_vector_type(16))) __bf16 v16bf;
typedef __attribute__((ext_vector_type(8)))  __bf16 v8bf;
typedef __attribute__((ext_vector_type(8)))  float  v8f;
typedef __attribute__((ext_vector_type(4)))  unsigned int u32x4;
typedef __attribute__((ext_vector_type(8)))  int    i32x8;
typedef __attribute__((ext_vector_type(4)))  int    i32x4;

// =============================== TDM helpers ===============================
__device__ __forceinline__ void tdm_issue(u32x4 g0, i32x8 g1) {
  i32x4 z4 = {0, 0, 0, 0};
#if __clang_major__ >= 23
  i32x8 z8 = {0, 0, 0, 0, 0, 0, 0, 0};
  __builtin_amdgcn_tensor_load_to_lds(g0, g1, z4, z4, z8, 0);
#else
  __builtin_amdgcn_tensor_load_to_lds(g0, g1, z4, z4, 0);
#endif
}

// 2D fp32 tile global -> LDS with per-row LDS padding (fused fallback path).
__device__ __forceinline__ void tdm_load_2d_f32(unsigned lds_addr, const float* gptr,
                                                unsigned tile_d0, unsigned tile_d1,
                                                unsigned tensor_d0, unsigned d0_stride,
                                                unsigned pad_interval_code,
                                                unsigned pad_amount_code) {
  unsigned long long ga = (unsigned long long)(uintptr_t)gptr;
  u32x4 g0;
  g0[0] = 1u;
  g0[1] = lds_addr;
  g0[2] = (unsigned)ga;
  g0[3] = (unsigned)((ga >> 32) & 0x01FFFFFFu) | (2u << 30);
  i32x8 g1;
  g1[0] = (int)((2u << 16) | (1u << 20) | (pad_interval_code << 22) | (pad_amount_code << 25));
  g1[1] = (int)((tensor_d0 & 0xFFFFu) << 16);
  g1[2] = (int)((tensor_d0 >> 16) | (0xFFFFu << 16));
  g1[3] = (int)(0xFu | (tile_d0 << 16));
  g1[4] = (int)(tile_d1 & 0xFFFFu);
  g1[5] = (int)d0_stride;
  g1[6] = 0;
  g1[7] = 0;
  tdm_issue(g0, g1);
}

// 1D bf16 blob global -> LDS (fast path; data lands fragment-major as stored).
__device__ __forceinline__ void tdm_load_1d_bf16(unsigned lds_addr, const void* gptr,
                                                 unsigned nelem) {
  unsigned long long ga = (unsigned long long)(uintptr_t)gptr;
  u32x4 g0;
  g0[0] = 1u;
  g0[1] = lds_addr;
  g0[2] = (unsigned)ga;
  g0[3] = (unsigned)((ga >> 32) & 0x01FFFFFFu) | (2u << 30);
  i32x8 g1;
  g1[0] = (int)(1u << 16);                    // data_size = 2 bytes, no pad
  g1[1] = (int)((nelem & 0xFFFFu) << 16);     // tensor_dim0 lo
  g1[2] = (int)((nelem >> 16) | (1u << 16));  // tensor_dim0 hi | tensor_dim1 = 1
  g1[3] = (int)(nelem << 16);                 // tile_dim0 = nelem (<= 65535)
  g1[4] = 1;                                  // tile_dim1 = 1
  g1[5] = (int)nelem;                         // dim0 stride (single row)
  g1[6] = 0;
  g1[7] = 0;
  tdm_issue(g0, g1);
}

// fp32 -> hi/lo bf16 split; f ~= hi + lo (residual via v_fma_mix_f32_bf16).
__device__ __forceinline__ void split_bf16(float f, __bf16& hi, __bf16& lo) {
  unsigned u  = __float_as_uint(f);
  unsigned hb = (u + 0x8000u) & 0xFFFF0000u;
  float    fh = __uint_as_float(hb);
  float    r  = f - fh;
  unsigned ur = __float_as_uint(r);
  hi = __builtin_bit_cast(__bf16, (unsigned short)(hb >> 16));
  lo = __builtin_bit_cast(__bf16, (unsigned short)((ur + 0x8000u) >> 16));
}

// ====================== phase 1a: x -> fragment-major bf16 ==================
// One block per (t, nchunk, b). Blob: [8 msub][32 lanes][16 e], A lane order:
// e<8 -> K=g*8+e ; e>=8 -> K=16+g*8+(e-8)   (ISA 7.12.2, 16-bit A 16x32)
__global__ __launch_bounds__(256) void prep_x_kernel(
    const float* __restrict__ x, unsigned short* __restrict__ xh_,
    unsigned short* __restrict__ xl_) {
  __bf16* xh = (__bf16*)xh_;
  __bf16* xl = (__bf16*)xl_;
  const int t   = blockIdx.x;
  const int nc  = blockIdx.y;
  const int b   = blockIdx.z;
  const int tid = threadIdx.x, lane = tid & 31;
  const int msub = tid >> 5;
  const int row  = nc * BM + msub * 16 + (lane & 15);
  const int g    = lane >> 4;
  const float* src = x + ((size_t)b * NSEQ + row) * INDIM + t * KB;
  v16bf vh, vl;
#pragma unroll
  for (int e = 0; e < 16; ++e) {
    const int k = (e < 8) ? (g * 8 + e) : (16 + g * 8 + (e - 8));
    __bf16 h, l;
    split_bf16(src[k], h, l);
    vh[e] = h; vl[e] = l;
  }
  const size_t off = (((size_t)(b * NCHUNKS + nc)) * KSTEPS + t) * XBLOB
                   + (size_t)(msub * 32 + lane) * 16;
  *(v16bf*)(xh + off) = vh;     // consecutive lanes -> consecutive 32B (coalesced)
  *(v16bf*)(xl + off) = vl;
}

// ====================== phase 1b: W[idx] -> fragment-major bf16 =============
// One block per (t, jchunk, b). Blob: [4 nsub][32 lanes][16 e], B lane order:
// e<8 -> K=g*8+e ; e>=8 -> K=16+g*8+(e-8)   (same order as A)
__global__ __launch_bounds__(256) void prep_w_kernel(
    const float* __restrict__ Wp, const long long* __restrict__ idx,
    unsigned short* __restrict__ wh_, unsigned short* __restrict__ wl_) {
  __bf16* wh = (__bf16*)wh_;
  __bf16* wl = (__bf16*)wl_;
  const int t   = blockIdx.x;
  const int jc  = blockIdx.y;
  const int b   = blockIdx.z;
  const int p   = (int)idx[b];
  const int tid = threadIdx.x;
  const int s   = tid & 127;
  const int nsub = s >> 5;
  const int wlan = s & 31;
  const int eb   = (tid >> 7) * 8;
  const int col  = jc * BN + nsub * 16 + (wlan & 15);
  const int k0   = t * KB + (wlan >> 4) * 8 + eb * 2;
  const float* src = Wp + (size_t)p * INDIM * OUTDIM;
  v8bf uh, ul;
#pragma unroll
  for (int e = 0; e < 8; ++e) {
    __bf16 h, l;
    split_bf16(src[(size_t)(k0 + e) * OUTDIM + col], h, l);
    uh[e] = h; ul[e] = l;
  }
  const size_t off = (((size_t)(b * JCHUNKS + jc)) * KSTEPS + t) * WBLOB
                   + (size_t)(nsub * 32 + wlan) * 16 + eb;
  *(v8bf*)(wh + off) = uh;
  *(v8bf*)(wl + off) = ul;
}

// ========================= phase 2: pure-WMMA GEMM ==========================
// TDM streams fragment-major bf16 blobs into double-buffered LDS; inner loop
// is s_wait_tensorcnt -> barrier -> ds_load_b128 fragments -> 12 WMMA.
__global__ __launch_bounds__(256) void gemm_bf16_kernel(
    const unsigned short* __restrict__ xh_, const unsigned short* __restrict__ xl_,
    const unsigned short* __restrict__ wh_, const unsigned short* __restrict__ wl_,
    const float* __restrict__ Bp, const long long* __restrict__ idx,
    float* __restrict__ out) {
  extern __shared__ char smem[];
  __bf16* sxh = (__bf16*)(smem + FOFF_XH);  // [2][XBLOB]
  __bf16* sxl = (__bf16*)(smem + FOFF_XL);
  __bf16* swh = (__bf16*)(smem + FOFF_WH);  // [2][WBLOB]
  __bf16* swl = (__bf16*)(smem + FOFF_WL);

  const int jc = blockIdx.x;        // 0..15
  const int nc = blockIdx.y;        // 0..3
  const int b  = blockIdx.z;
  const int p  = (int)idx[b];

  const int tid  = threadIdx.x;
  const int lane = tid & 31;
  const int wid  = tid >> 5;
  const int mw   = wid & 3;         // 4 row-waves (32 rows each)
  const int nw   = wid >> 2;        // 2 col-waves (32 cols each)

  const __bf16* gxh = (const __bf16*)xh_ + ((size_t)(b * NCHUNKS + nc)) * KSTEPS * XBLOB;
  const __bf16* gxl = (const __bf16*)xl_ + ((size_t)(b * NCHUNKS + nc)) * KSTEPS * XBLOB;
  const __bf16* gwh = (const __bf16*)wh_ + ((size_t)(b * JCHUNKS + jc)) * KSTEPS * WBLOB;
  const __bf16* gwl = (const __bf16*)wl_ + ((size_t)(b * JCHUNKS + jc)) * KSTEPS * WBLOB;

  // Prologue: stage K-step 0 into buffer 0.
  if (wid == 0) {
    tdm_load_1d_bf16((unsigned)(uintptr_t)(void*)(sxh), gxh, XBLOB);
    tdm_load_1d_bf16((unsigned)(uintptr_t)(void*)(sxl), gxl, XBLOB);
    tdm_load_1d_bf16((unsigned)(uintptr_t)(void*)(swh), gwh, WBLOB);
    tdm_load_1d_bf16((unsigned)(uintptr_t)(void*)(swl), gwl, WBLOB);
  }

  v8f acc[2][2] = {};

  for (int t = 0; t < KSTEPS; ++t) {
    const int cur = t & 1;
    if (wid == 0) __builtin_amdgcn_s_wait_tensorcnt(0);  // buffer cur landed
    __syncthreads();  // all waves done reading buffer cur^1 (prev iter), data ready

    if (wid == 0 && t + 1 < KSTEPS) {   // prefetch next K-step (overlaps compute)
      const int nb = cur ^ 1;
      tdm_load_1d_bf16((unsigned)(uintptr_t)(void*)(sxh + nb * XBLOB),
                       gxh + (size_t)(t + 1) * XBLOB, XBLOB);
      tdm_load_1d_bf16((unsigned)(uintptr_t)(void*)(sxl + nb * XBLOB),
                       gxl + (size_t)(t + 1) * XBLOB, XBLOB);
      tdm_load_1d_bf16((unsigned)(uintptr_t)(void*)(swh + nb * WBLOB),
                       gwh + (size_t)(t + 1) * WBLOB, WBLOB);
      tdm_load_1d_bf16((unsigned)(uintptr_t)(void*)(swl + nb * WBLOB),
                       gwl + (size_t)(t + 1) * WBLOB, WBLOB);
    }

    const v16bf* XH = (const v16bf*)(sxh + cur * XBLOB);
    const v16bf* XL = (const v16bf*)(sxl + cur * XBLOB);
    const v16bf* WH = (const v16bf*)(swh + cur * WBLOB);
    const v16bf* WL = (const v16bf*)(swl + cur * WBLOB);
    v16bf Ah[2], Al[2], Bh[2], Bl[2];
#pragma unroll
    for (int ms = 0; ms < 2; ++ms) {
      Ah[ms] = XH[(mw * 2 + ms) * 32 + lane];
      Al[ms] = XL[(mw * 2 + ms) * 32 + lane];
    }
#pragma unroll
    for (int ns = 0; ns < 2; ++ns) {
      Bh[ns] = WH[(nw * 2 + ns) * 32 + lane];
      Bl[ns] = WL[(nw * 2 + ns) * 32 + lane];
    }

    // Error-compensated fp32 GEMM: small terms first, then hi*hi.
#pragma unroll
    for (int ms = 0; ms < 2; ++ms)
#pragma unroll
      for (int ns = 0; ns < 2; ++ns) {
        acc[ms][ns] = __builtin_amdgcn_wmma_f32_16x16x32_bf16(
            false, Al[ms], false, Bh[ns], (short)0, acc[ms][ns], false, false);
        acc[ms][ns] = __builtin_amdgcn_wmma_f32_16x16x32_bf16(
            false, Ah[ms], false, Bl[ns], (short)0, acc[ms][ns], false, false);
        acc[ms][ns] = __builtin_amdgcn_wmma_f32_16x16x32_bf16(
            false, Ah[ms], false, Bh[ns], (short)0, acc[ms][ns], false, false);
      }
  }

  // Epilogue: C/D layout -> M = vgpr + 8*(lane>=16), N = lane&15. Bias + store.
  const int col = lane & 15;
  const int r0  = (lane >> 4) * 8;
#pragma unroll
  for (int ns = 0; ns < 2; ++ns) {
    const int jcg = jc * BN + nw * 32 + ns * 16 + col;
    const float bias = Bp[(size_t)p * OUTDIM + jcg];
#pragma unroll
    for (int ms = 0; ms < 2; ++ms) {
      const int rbase = nc * BM + mw * 32 + ms * 16 + r0;
#pragma unroll
      for (int r = 0; r < 8; ++r) {
        out[((size_t)b * NSEQ + rbase + r) * OUTDIM + jcg] = acc[ms][ns][r] + bias;
      }
    }
  }
}

// ================= fallback: fused kernel (round-2) =============
__global__ __launch_bounds__(256) void l2m_fused_kernel(
    const float* __restrict__ x, const float* __restrict__ Wp,
    const float* __restrict__ Bp, const long long* __restrict__ idx,
    float* __restrict__ out) {
  extern __shared__ char smem[];
  float*  sx = (float*)(smem + OFF_SX);
  float*  sw = (float*)(smem + OFF_SW);
  __bf16* xh = (__bf16*)(smem + OFF_XH);
  __bf16* xl = (__bf16*)(smem + OFF_XL);
  __bf16* wh = (__bf16*)(smem + OFF_WH);
  __bf16* wl = (__bf16*)(smem + OFF_WL);

  const int j0 = blockIdx.x * BN;
  const int n0 = blockIdx.y * BM;
  const int b  = blockIdx.z;
  const int p  = (int)idx[b];

  const int tid  = threadIdx.x;
  const int lane = tid & 31;
  const int wid  = tid >> 5;
  const int mw   = wid & 3;
  const int nw   = wid >> 2;

  const float* xbase = x  + ((size_t)b * NSEQ + n0) * INDIM;
  const float* wbase = Wp + (size_t)p * INDIM * OUTDIM + j0;

  if (wid == 0) {
    tdm_load_2d_f32((unsigned)(uintptr_t)(void*)sx, xbase, KB, BM, INDIM, INDIM, 4u, 3u);
    tdm_load_2d_f32((unsigned)(uintptr_t)(void*)sw, wbase, BN, KB, OUTDIM, OUTDIM, 5u, 1u);
  }

  const int xsub  = tid >> 5;
  const int xrow  = xsub * 16 + (lane & 15);
  const int xg    = lane >> 4;
  const int ws_   = tid & 127;
  const int wsub  = ws_ >> 5;
  const int wlane = ws_ & 31;
  const int web   = (tid >> 7) * 8;
  const int wcol  = wsub * 16 + (wlane & 15);
  const int wrow0 = (wlane >> 4) * 8 + web * 2;

  v8f acc[2][2] = {};

  for (int t = 0; t < KSTEPS; ++t) {
    const int cur = t & 1;
    if (wid == 0) {
      if (t + 1 < KSTEPS) {
        const int k1 = (t + 1) * KB;
        tdm_load_2d_f32((unsigned)(uintptr_t)(void*)(sx + (cur ^ 1) * (BM * XPITCH)),
                        xbase + k1, KB, BM, INDIM, INDIM, 4u, 3u);
        tdm_load_2d_f32((unsigned)(uintptr_t)(void*)(sw + (cur ^ 1) * (KB * WPITCH)),
                        wbase + (size_t)k1 * OUTDIM, BN, KB, OUTDIM, OUTDIM, 5u, 1u);
        __builtin_amdgcn_s_wait_tensorcnt(2);
      } else {
        __builtin_amdgcn_s_wait_tensorcnt(0);
      }
    }
    __syncthreads();

    {
      const float* xsrc = sx + cur * (BM * XPITCH) + xrow * XPITCH;
      v16bf vh, vlv;
#pragma unroll
      for (int e = 0; e < 16; ++e) {
        const int k = (e < 8) ? (xg * 8 + e) : (16 + xg * 8 + (e - 8));
        __bf16 h, l;
        split_bf16(xsrc[k], h, l);
        vh[e] = h; vlv[e] = l;
      }
      *(v16bf*)(xh + (xsub * 32 + (tid & 31)) * 16) = vh;
      *(v16bf*)(xl + (xsub * 32 + (tid & 31)) * 16) = vlv;

      const float* wsrc = sw + cur * (KB * WPITCH);
      v8bf uh, ul;
#pragma unroll
      for (int e = 0; e < 8; ++e) {
        __bf16 h, l;
        split_bf16(wsrc[(wrow0 + e) * WPITCH + wcol], h, l);
        uh[e] = h; ul[e] = l;
      }
      *(v8bf*)(wh + (wsub * 32 + wlane) * 16 + web) = uh;
      *(v8bf*)(wl + (wsub * 32 + wlane) * 16 + web) = ul;
    }
    __syncthreads();

    const v16bf* XH = (const v16bf*)xh;
    const v16bf* XL = (const v16bf*)xl;
    const v16bf* WH = (const v16bf*)wh;
    const v16bf* WL = (const v16bf*)wl;
    v16bf Ah[2], Al[2], Bh[2], Bl[2];
#pragma unroll
    for (int ms = 0; ms < 2; ++ms) {
      Ah[ms] = XH[(mw * 2 + ms) * 32 + lane];
      Al[ms] = XL[(mw * 2 + ms) * 32 + lane];
    }
#pragma unroll
    for (int ns = 0; ns < 2; ++ns) {
      Bh[ns] = WH[(nw * 2 + ns) * 32 + lane];
      Bl[ns] = WL[(nw * 2 + ns) * 32 + lane];
    }
#pragma unroll
    for (int ms = 0; ms < 2; ++ms)
#pragma unroll
      for (int ns = 0; ns < 2; ++ns) {
        acc[ms][ns] = __builtin_amdgcn_wmma_f32_16x16x32_bf16(
            false, Al[ms], false, Bh[ns], (short)0, acc[ms][ns], false, false);
        acc[ms][ns] = __builtin_amdgcn_wmma_f32_16x16x32_bf16(
            false, Ah[ms], false, Bl[ns], (short)0, acc[ms][ns], false, false);
        acc[ms][ns] = __builtin_amdgcn_wmma_f32_16x16x32_bf16(
            false, Ah[ms], false, Bh[ns], (short)0, acc[ms][ns], false, false);
      }
  }

  const int col = lane & 15;
  const int r0  = (lane >> 4) * 8;
#pragma unroll
  for (int ns = 0; ns < 2; ++ns) {
    const int jc = j0 + nw * 32 + ns * 16 + col;
    const float bias = Bp[(size_t)p * OUTDIM + jc];
#pragma unroll
    for (int ms = 0; ms < 2; ++ms) {
      const int rbase = n0 + mw * 32 + ms * 16 + r0;
#pragma unroll
      for (int r = 0; r < 8; ++r) {
        out[((size_t)b * NSEQ + rbase + r) * OUTDIM + jc] = acc[ms][ns][r] + bias;
      }
    }
  }
}

extern "C" void kernel_launch(void* const* d_in, const int* in_sizes, int n_in,
                              void* d_out, int out_size, void* d_ws, size_t ws_size,
                              hipStream_t stream) {
  const float*     xp  = (const float*)d_in[0];
  const float*     Wp  = (const float*)d_in[1];
  const float*     Bp  = (const float*)d_in[2];
  const long long* idx = (const long long*)d_in[3];
  // d_in[4] = frozen_mask: gradients only; forward output identical.
  float* out = (float*)d_out;

  if (ws_size >= WS_NEED) {
    unsigned short* xh = (unsigned short*)((char*)d_ws + WS_XH);
    unsigned short* xl = (unsigned short*)((char*)d_ws + WS_XL);
    unsigned short* wh = (unsigned short*)((char*)d_ws + WS_WH);
    unsigned short* wl = (unsigned short*)((char*)d_ws + WS_WL);
    hipLaunchKernelGGL(prep_x_kernel, dim3(KSTEPS, NCHUNKS, BATCH), dim3(256), 0,
                       stream, xp, xh, xl);
    hipLaunchKernelGGL(prep_w_kernel, dim3(KSTEPS, JCHUNKS, BATCH), dim3(256), 0,
                       stream, Wp, idx, wh, wl);
    hipLaunchKernelGGL(gemm_bf16_kernel, dim3(JCHUNKS, NCHUNKS, BATCH), dim3(256),
                       SMEM_FAST, stream, xh, xl, wh, wl, Bp, idx, out);
  } else {
    hipLaunchKernelGGL(l2m_fused_kernel, dim3(JCHUNKS, NCHUNKS, BATCH), dim3(256),
                       SMEM_FUSED, stream, xp, Wp, Bp, idx, out);
  }

  (void)in_sizes; (void)n_in; (void)out_size;
}
